// LSTMCell_21002390078075
// MI455X (gfx1250) — matrix operand order repaired and checked by reading, not compiled, not run on hardware
//
#include <hip/hip_runtime.h>

typedef __attribute__((ext_vector_type(16))) _Float16 v16h;
typedef __attribute__((ext_vector_type(8)))  _Float16 v8h;
typedef __attribute__((ext_vector_type(4)))  _Float16 v4h;
typedef __attribute__((ext_vector_type(8)))  float    v8f;

#define HDIM 512
#define BM   128          // rows per workgroup
#define LDSS 520          // LDS row stride in halfs (512 + 8 pad -> bank spread)
#define NBLK (HDIM / 16)  // 32 n-blocks of 16
#define KBLK (HDIM / 32)  // 16 k-blocks of 32

__device__ __forceinline__ float sigm_(float x) {
    return 1.0f / (1.0f + __expf(-x));
}
__device__ __forceinline__ float tanh_(float x) {
    float e = __expf(-2.0f * fabsf(x));
    float r = (1.0f - e) / (1.0f + e);
    return copysignf(r, x);
}

// ---------------------------------------------------------------------------
// Pack Wf/Wi/Wg/Wo (fp32, row-major [n][k], used as W^T) into f16 WMMA
// B-operand fragment order:
//   fragment(g, nb, kb) = 32 lanes x 16 halfs, lane = ((k%16)/8)*16 + n%16,
//   half j within lane = (k%8) + 8*((k%32)/16)  (lanes 0-15 hold K=0..7,16..23,
//   lanes 16-31 hold K=8..15,24..31, one column per lane).
// ---------------------------------------------------------------------------
__global__ void pack_weights_f16(const float* __restrict__ Wf,
                                 const float* __restrict__ Wi,
                                 const float* __restrict__ Wg,
                                 const float* __restrict__ Wo,
                                 _Float16* __restrict__ out) {
    int i = blockIdx.x * 256 + threadIdx.x;          // 0 .. 4*512*512-1
    if (i >= 4 * HDIM * HDIM) return;
    int g   = i >> 18;
    int rem = i & ((1 << 18) - 1);
    int n   = rem >> 9;
    int k   = rem & (HDIM - 1);
    const float* W = (g == 0) ? Wf : (g == 1) ? Wi : (g == 2) ? Wg : Wo;
    float v = W[n * HDIM + k];
    int nb   = n >> 4;
    int nl   = n & 15;
    int kb   = k >> 5;
    int kg   = (k >> 3) & 1;     // lane half-group
    int j    = (k & 7) + 8 * ((k >> 4) & 1);
    int lane = kg * 16 + nl;
    size_t idx = ((((size_t)(g * NBLK + nb) * KBLK) + kb) * 32 + lane) * 16 + j;
    out[idx] = (_Float16)v;
}

// ---------------------------------------------------------------------------
// Fused LSTM-cell kernel: z = x + stm in LDS (f16), 4 gate GEMMs via
// v_wmma_f32_16x16x32_f16 sharing A-fragments, fused gate nonlinearities.
// Block: 256 threads (8 waves). Block computes rows [blockIdx*128, +128) x
// all 512 columns. Wave tile: 32x32, 8 column tiles per wave.
// ---------------------------------------------------------------------------
__global__ __launch_bounds__(256) void lstm_cell_wmma(
    const float* __restrict__ x,   const float* __restrict__ stm,
    const _Float16* __restrict__ Wp,
    const float* __restrict__ bf,  const float* __restrict__ bi,
    const float* __restrict__ bg,  const float* __restrict__ bo,
    float* __restrict__ outC,      float* __restrict__ outH) {
    extern __shared__ _Float16 zt[];   // [BM][LDSS] = 133,120 bytes

    const int tid  = threadIdx.x;
    const int wave = tid >> 5;
    const int lane = tid & 31;
    const int R0   = blockIdx.x * BM;

    // ---- Phase 1: z tile -> LDS (f16), float4-vectorized -------------------
    for (int i = tid; i < BM * HDIM / 4; i += 256) {
        int idx = i * 4;
        int r   = idx >> 9;          // 0..127
        int c   = idx & (HDIM - 1);  // multiple of 4
        const float4 a = *reinterpret_cast<const float4*>(&x[(size_t)(R0 + r) * HDIM + c]);
        const float4 b = *reinterpret_cast<const float4*>(&stm[(size_t)(R0 + r) * HDIM + c]);
        v4h z4;
        z4[0] = (_Float16)(a.x + b.x);
        z4[1] = (_Float16)(a.y + b.y);
        z4[2] = (_Float16)(a.z + b.z);
        z4[3] = (_Float16)(a.w + b.w);
        *reinterpret_cast<v4h*>(&zt[r * LDSS + c]) = z4;
    }
    __syncthreads();

    // ---- Phase 2: WMMA tiles ----------------------------------------------
    const int m0 = (wave & 3) * 32;        // row block within workgroup
    const int ll = lane & 15;              // lane-in-halfwave
    const int lh = lane >> 4;              // half-wave select

    for (int t = 0; t < 8; ++t) {
        const int cb = (wave >> 2) + 2 * t;    // 0..15 (32-col block)
        const int n0 = cb * 32;

        v8f acc[4][2][2] = {};                 // [gate][mi][ni]

        for (int kb = 0; kb < KBLK; ++kb) {
            const int k0 = kb * 32;

            // A fragments: 16-bit A 16x32 layout. lane row = m + (lane&15),
            // chunk (lane>>4): halfs k0+c*8..+7 (v0-3) and k0+16+c*8 (v4-7).
            v16h afrag[2];
#pragma unroll
            for (int mi = 0; mi < 2; ++mi) {
                const _Float16* p =
                    &zt[(m0 + 16 * mi + ll) * LDSS + k0 + lh * 8];
                v8h lo = *reinterpret_cast<const v8h*>(p);
                v8h hi = *reinterpret_cast<const v8h*>(p + 16);
                afrag[mi] = __builtin_shufflevector(
                    lo, hi, 0, 1, 2, 3, 4, 5, 6, 7, 8, 9, 10, 11, 12, 13, 14, 15);
            }

#pragma unroll
            for (int g = 0; g < 4; ++g) {
#pragma unroll
                for (int ni = 0; ni < 2; ++ni) {
                    const int nb = cb * 2 + ni;
                    const _Float16* bp =
                        Wp + ((((size_t)(g * NBLK + nb) * KBLK) + kb) * 32 + lane) * 16;
                    v16h bfrag = *reinterpret_cast<const v16h*>(bp);
#pragma unroll
                    for (int mi = 0; mi < 2; ++mi) {
                        acc[g][mi][ni] = __builtin_amdgcn_wmma_f32_16x16x32_f16(
                            false, afrag[mi], false, bfrag,
                            (short)0, acc[g][mi][ni], false, false);
                    }
                }
            }
        }

        // ---- Epilogue: gate combine + store (C/D layout: n = lane&15,
        // m = vgpr + 8*(lane>>4)) ------------------------------------------
#pragma unroll
        for (int mi = 0; mi < 2; ++mi) {
#pragma unroll
            for (int ni = 0; ni < 2; ++ni) {
                const int n   = n0 + ni * 16 + ll;
                const float bfv = bf[n];
                const float biv = bi[n];
                const float bgv = bg[n];
                const float bov = bo[n];
                const int mbase = R0 + m0 + 16 * mi + lh * 8;
#pragma unroll
                for (int r = 0; r < 8; ++r) {
                    float fg = sigm_(acc[0][mi][ni][r] + bfv);
                    float ig = sigm_(acc[1][mi][ni][r] + biv);
                    float gg = tanh_(acc[2][mi][ni][r] + bgv);
                    float og = sigm_(acc[3][mi][ni][r] + bov);
                    float c  = fg + ig * gg;
                    float h  = tanh_(c) * og;
                    size_t o = (size_t)(mbase + r) * HDIM + n;
                    outC[o] = c;
                    outH[o] = h;
                }
            }
        }
    }
}

extern "C" void kernel_launch(void* const* d_in, const int* in_sizes, int n_in,
                              void* d_out, int out_size, void* d_ws, size_t ws_size,
                              hipStream_t stream) {
    const float* x   = (const float*)d_in[0];
    const float* stm = (const float*)d_in[1];
    const float* Wf  = (const float*)d_in[2];
    const float* bf  = (const float*)d_in[3];
    const float* Wi  = (const float*)d_in[4];
    const float* bi  = (const float*)d_in[5];
    const float* Wg  = (const float*)d_in[6];
    const float* bg  = (const float*)d_in[7];
    const float* Wo  = (const float*)d_in[8];
    const float* bo  = (const float*)d_in[9];

    _Float16* Wp = (_Float16*)d_ws;                    // 4*512*512*2 = 2 MB
    float* outC  = (float*)d_out;                      // [B, H]
    float* outH  = (float*)d_out + (size_t)65536 * HDIM;

    // 1) pack weights to f16 WMMA fragment layout (stays hot in 192 MB L2)
    pack_weights_f16<<<(4 * HDIM * HDIM + 255) / 256, 256, 0, stream>>>(
        Wf, Wi, Wg, Wo, Wp);

    // 2) fused z + 4-gate WMMA GEMM + nonlinearities
    const size_t lds_bytes = (size_t)BM * LDSS * sizeof(_Float16);  // 133,120
    lstm_cell_wmma<<<65536 / BM, 256, lds_bytes, stream>>>(
        x, stm, Wp, bf, bi, bg, bo, outC, outH);
}
